// GenerateLocalHypergraph_69501160784166
// MI455X (gfx1250) — matrix operand order repaired and checked by reading, not compiled, not run on hardware
//
#include <hip/hip_runtime.h>
#include <hip/hip_bf16.h>

// Shapes from the reference
#define N_  64
#define T_  64
#define F_  64
#define KN_ 64
#define E_  128
#define TN_ 4096

typedef __attribute__((ext_vector_type(16))) __bf16 v16bf;
typedef __attribute__((ext_vector_type(8)))  float  v8f;

struct alignas(16) U4 { unsigned int x[4]; };
union FragBF { v16bf v; U4 q[2]; };

// ---------------- elementwise / setup kernels ----------------

__global__ void k_cvt_bf16(const float* __restrict__ in, __bf16* __restrict__ out, int n) {
  for (int i = blockIdx.x * blockDim.x + threadIdx.x; i < n; i += gridDim.x * blockDim.x)
    out[i] = (__bf16)in[i];
}

// xf[t*N+n, f] = x[n, t, f]
__global__ void k_xf(const float* __restrict__ x, float* __restrict__ xf) {
  int gid = blockIdx.x * 256 + threadIdx.x;   // 262144 total
  int i = gid >> 6, f = gid & 63;
  int t = i >> 6, n = i & 63;
  xf[gid] = x[(n * T_ + t) * F_ + f];
}

// q = xf@Wq^T + bq ; k = xf@Wk^T + bk   (Wq = rows 0..63, Wk = rows 64..127 of in_proj_w)
__global__ void k_qk(const float* __restrict__ xf, const float* __restrict__ w,
                     const float* __restrict__ b, float* __restrict__ q, float* __restrict__ k) {
  int gid = blockIdx.x * 256 + threadIdx.x;
  int i = gid >> 6, f = gid & 63;
  const float* xr = xf + i * F_;
  const float* wq = w + f * F_;
  const float* wk = w + (F_ + f) * F_;
  float aq = b[f], ak = b[F_ + f];
  for (int c = 0; c < F_; ++c) { float xv = xr[c]; aq += xv * wq[c]; ak += xv * wk[c]; }
  q[gid] = aq; k[gid] = ak;
}

// ---------------- attention: scores + mask + softmax (one WG per row) ----------------

__global__ void k_attn(const float* __restrict__ q, const float* __restrict__ kmat,
                       float* __restrict__ attn) {
  int i = blockIdx.x;
  int tid = threadIdx.x;
  int limit = ((i >> 6) + 1) << 6;       // block-causal: t_j <= t_i
  __shared__ float qs[F_];
  __shared__ float red[256];
  if (tid < F_) qs[tid] = q[i * F_ + tid];
  __syncthreads();
  float sloc[16];
  int cnt = 0;
  float lmax = -3.0e38f;
  for (int j = tid; j < limit; j += 256) {
    const float* kr = kmat + j * F_;
    float s = 0.f;
    for (int c = 0; c < F_; ++c) s += qs[c] * kr[c];
    s *= 0.125f;                          // / sqrt(64)
    sloc[cnt++] = s;
    lmax = fmaxf(lmax, s);
  }
  red[tid] = lmax; __syncthreads();
  for (int s = 128; s > 0; s >>= 1) { if (tid < s) red[tid] = fmaxf(red[tid], red[tid + s]); __syncthreads(); }
  float m = red[0]; __syncthreads();
  float lsum = 0.f; cnt = 0;
  for (int j = tid; j < limit; j += 256) { float e = __expf(sloc[cnt] - m); sloc[cnt] = e; lsum += e; ++cnt; }
  red[tid] = lsum; __syncthreads();
  for (int s = 128; s > 0; s >>= 1) { if (tid < s) red[tid] += red[tid + s]; __syncthreads(); }
  float inv = 1.f / red[0];
  float* arow = attn + (size_t)i * TN_;
  cnt = 0;
  for (int j = tid; j < limit; j += 256) arow[j] = sloc[cnt++] * inv;
  for (int j = limit + tid; j < TN_; j += 256) arow[j] = 0.f;   // masked -> softmax gives 0
}

// ---------------- top-64 per row + re-softmax over kept values ----------------

__global__ void k_topk(float* __restrict__ attn, __bf16* __restrict__ atkb) {
  int i = blockIdx.x, tid = threadIdx.x;
  int limit = ((i >> 6) + 1) << 6;
  __shared__ float vals[TN_];
  __shared__ float rv[256];
  __shared__ int   ri[256];
  __shared__ int   selIdx[KN_];
  __shared__ float selVal[KN_];
  float* arow = attn + (size_t)i * TN_;
  for (int j = tid; j < limit; j += 256) vals[j] = arow[j];
  __syncthreads();
  for (int it = 0; it < KN_; ++it) {
    // local argmax (lowest index wins on ties since we scan ascending)
    float bv = -2.f; int bi = 1 << 30;
    for (int j = tid; j < limit; j += 256) {
      float v = vals[j];
      if (v > bv) { bv = v; bi = j; }
    }
    rv[tid] = bv; ri[tid] = bi; __syncthreads();
    for (int s = 128; s > 0; s >>= 1) {
      if (tid < s) {
        float v2 = rv[tid + s]; int i2 = ri[tid + s];
        if (v2 > rv[tid] || (v2 == rv[tid] && i2 < ri[tid])) { rv[tid] = v2; ri[tid] = i2; }
      }
      __syncthreads();
    }
    if (tid == 0) { selIdx[it] = ri[0]; selVal[it] = rv[0]; vals[ri[0]] = -1.f; }
    __syncthreads();
  }
  // softmax over the 64 kept attn values (iteration 0 selected the max)
  float mv = selVal[0];
  rv[tid] = (tid < KN_) ? __expf(selVal[tid] - mv) : 0.f;
  __syncthreads();
  for (int s = 128; s > 0; s >>= 1) { if (tid < s) rv[tid] += rv[tid + s]; __syncthreads(); }
  float inv = 1.f / rv[0];
  __syncthreads();
  __bf16* brow = atkb + (size_t)i * TN_;
  for (int j = tid; j < TN_; j += 256) { arow[j] = 0.f; brow[j] = (__bf16)0.f; }
  __syncthreads();
  if (tid < KN_) {
    float v = __expf(selVal[tid] - mv) * inv;
    arow[selIdx[tid]] = v;
    brow[selIdx[tid]] = (__bf16)v;
  }
}

// ---------------- WMMA GEMM: C = tanh(relu(A @ B^T + bias)) ------------------
// A:[4096,K] bf16 row-major, B:[Ncols,K] bf16 row-major.
// Register-blocked: one wave computes a 32x32 output tile = 4 accumulators.
// Per K-step (32): 2 A-frags + 2 B-frags (8 x b128 loads) feed 4 WMMAs.
// A frag (16x32, 16-bit layout): lane<16 holds M=lane, K {0..7,16..23}; lane>=16: K {8..15,24..31}.
// B frag (32x16): lane<16 holds N=lane, K {0..7,16..23}; lane>=16: K {8..15,24..31}.
// Acc: VGPR r of lane -> row r + 8*(lane>=16), col = lane&15.

#define WMMA_BF16(A_, B_, C_) \
  __builtin_amdgcn_wmma_f32_16x16x32_bf16(false, (A_), false, (B_), (short)0, (C_), false, false)

__global__ void k_wmma_gemm(const __bf16* __restrict__ A, const __bf16* __restrict__ B,
                            const float* __restrict__ bias, __bf16* __restrict__ outBf,
                            float* __restrict__ outF, int Ncols, int K) {
  int wid  = threadIdx.x >> 5;
  int lane = threadIdx.x & 31;
  int m0 = blockIdx.x * 32;
  int n0 = (blockIdx.y * 8 + wid) * 32;
  if (n0 >= Ncols) return;
  int r = lane & 15, hi = lane >> 4;
  const __bf16* a0 = A + (size_t)(m0 + r) * K;
  const __bf16* a1 = A + (size_t)(m0 + 16 + r) * K;
  const __bf16* b0 = B + (size_t)(n0 + r) * K;
  const __bf16* b1 = B + (size_t)(n0 + 16 + r) * K;
  v8f acc00 = {}, acc01 = {}, acc10 = {}, acc11 = {};
  for (int kk = 0; kk < K; kk += 32) {
    FragBF fa0, fa1, fb0, fb1;
    fa0.q[0] = *(const U4*)(a0 + kk + hi * 8);
    fa0.q[1] = *(const U4*)(a0 + kk + 16 + hi * 8);
    fa1.q[0] = *(const U4*)(a1 + kk + hi * 8);
    fa1.q[1] = *(const U4*)(a1 + kk + 16 + hi * 8);
    fb0.q[0] = *(const U4*)(b0 + kk + hi * 8);
    fb0.q[1] = *(const U4*)(b0 + kk + 16 + hi * 8);
    fb1.q[0] = *(const U4*)(b1 + kk + hi * 8);
    fb1.q[1] = *(const U4*)(b1 + kk + 16 + hi * 8);
    acc00 = WMMA_BF16(fa0.v, fb0.v, acc00);
    acc01 = WMMA_BF16(fa0.v, fb1.v, acc01);
    acc10 = WMMA_BF16(fa1.v, fb0.v, acc10);
    acc11 = WMMA_BF16(fa1.v, fb1.v, acc11);
  }
  int c0 = n0 + r, c1 = n0 + 16 + r;
  float bc0 = bias[c0], bc1 = bias[c1];
  for (int rr = 0; rr < 8; ++rr) {
    int row0 = m0 + hi * 8 + rr;
    int row1 = row0 + 16;
    float v00 = tanhf(fmaxf(acc00[rr] + bc0, 0.f));
    float v01 = tanhf(fmaxf(acc01[rr] + bc1, 0.f));
    float v10 = tanhf(fmaxf(acc10[rr] + bc0, 0.f));
    float v11 = tanhf(fmaxf(acc11[rr] + bc1, 0.f));
    if (outBf) {
      outBf[(size_t)row0 * Ncols + c0] = (__bf16)v00;
      outBf[(size_t)row0 * Ncols + c1] = (__bf16)v01;
      outBf[(size_t)row1 * Ncols + c0] = (__bf16)v10;
      outBf[(size_t)row1 * Ncols + c1] = (__bf16)v11;
    }
    if (outF) {
      outF[(size_t)row0 * Ncols + c0] = v00;
      outF[(size_t)row0 * Ncols + c1] = v01;
      outF[(size_t)row1 * Ncols + c0] = v10;
      outF[(size_t)row1 * Ncols + c1] = v11;
    }
  }
}

// ---------------- epilogue kernels ----------------

// ln = layernorm(xf) ; Y = ln @ P    (one WG of 64 threads per row)
__global__ void k_lnY(const float* __restrict__ xf, const float* __restrict__ g,
                      const float* __restrict__ bb, const float* __restrict__ P,
                      float* __restrict__ Y) {
  int i = blockIdx.x, tid = threadIdx.x;
  __shared__ float row[F_];
  __shared__ float red[F_];
  float v = xf[i * F_ + tid];
  red[tid] = v; __syncthreads();
  for (int s = 32; s > 0; s >>= 1) { if (tid < s) red[tid] += red[tid + s]; __syncthreads(); }
  float mu = red[0] * (1.f / F_); __syncthreads();
  float d = v - mu;
  red[tid] = d * d; __syncthreads();
  for (int s = 32; s > 0; s >>= 1) { if (tid < s) red[tid] += red[tid + s]; __syncthreads(); }
  float var = red[0] * (1.f / F_);
  float ln = d * rsqrtf(var + 1e-5f) * g[tid] + bb[tid];
  row[tid] = ln; __syncthreads();
  float acc = 0.f;
  for (int c = 0; c < F_; ++c) acc += row[c] * P[c * F_ + tid];
  Y[i * F_ + tid] = acc;
}

// Z = H^T @ Y   (128 x 64, reduction over 4096)
__global__ void k_Z(const float* __restrict__ H, const float* __restrict__ Y, float* __restrict__ Z) {
  int e = blockIdx.x, f = threadIdx.x;
  float acc = 0.f;
  for (int i = 0; i < TN_; ++i) acc += H[(size_t)i * E_ + e] * Y[i * F_ + f];
  Z[e * F_ + f] = acc;
}

// out = elu(H @ Z) ; xout[n,t,f] = xf[t*N+n,f] + out
__global__ void k_final(const float* __restrict__ H, const float* __restrict__ Z,
                        const float* __restrict__ xf, float* __restrict__ xout) {
  int gid = blockIdx.x * 256 + threadIdx.x;
  int i = gid >> 6, f = gid & 63;
  const float* hr = H + (size_t)i * E_;
  float acc = 0.f;
  for (int e = 0; e < E_; ++e) acc += hr[e] * Z[e * F_ + f];
  float o = acc > 0.f ? acc : (__expf(acc) - 1.f);   // elu, alpha=1
  int t = i >> 6, n = i & 63;
  xout[(n * T_ + t) * F_ + f] = xf[gid] + o;
}

// ---------------- launcher ----------------

extern "C" void kernel_launch(void* const* d_in, const int* in_sizes, int n_in,
                              void* d_out, int out_size, void* d_ws, size_t ws_size,
                              hipStream_t stream) {
  (void)in_sizes; (void)n_in; (void)out_size; (void)ws_size;
  const float* x   = (const float*)d_in[0];
  const float* ipw = (const float*)d_in[1];
  const float* ipb = (const float*)d_in[2];
  const float* W1  = (const float*)d_in[3];
  const float* b1  = (const float*)d_in[4];
  const float* W2  = (const float*)d_in[5];
  const float* b2  = (const float*)d_in[6];
  const float* P   = (const float*)d_in[7];
  const float* lng = (const float*)d_in[8];
  const float* lnb = (const float*)d_in[9];

  float* out  = (float*)d_out;                       // (xout | attn_topk | H) concat
  float* xout = out;
  float* attn = out + (size_t)N_ * T_ * F_;
  float* Hout = attn + (size_t)TN_ * TN_;

  char* ws = (char*)d_ws;
  size_t off = 0;
  auto alloc = [&](size_t bytes) -> void* {
    void* p = ws + off;
    off = (off + bytes + 255) & ~(size_t)255;
    return p;
  };
  float*  xf   = (float*)alloc((size_t)TN_ * F_ * 4);
  float*  q    = (float*)alloc((size_t)TN_ * F_ * 4);
  float*  kbuf = (float*)alloc((size_t)TN_ * F_ * 4);
  float*  Y    = (float*)alloc((size_t)TN_ * F_ * 4);
  float*  Z    = (float*)alloc((size_t)E_ * F_ * 4);
  __bf16* w1b  = (__bf16*)alloc((size_t)TN_ * TN_ * 2);
  __bf16* w2b  = (__bf16*)alloc((size_t)E_ * TN_ * 2);
  __bf16* atkb = (__bf16*)alloc((size_t)TN_ * TN_ * 2);
  __bf16* h1b  = (__bf16*)alloc((size_t)TN_ * TN_ * 2);

  hipLaunchKernelGGL(k_cvt_bf16, dim3(4096), dim3(256), 0, stream, W1, w1b, TN_ * TN_);
  hipLaunchKernelGGL(k_cvt_bf16, dim3(512),  dim3(256), 0, stream, W2, w2b, E_ * TN_);
  hipLaunchKernelGGL(k_xf,  dim3(1024), dim3(256), 0, stream, x, xf);
  hipLaunchKernelGGL(k_qk,  dim3(1024), dim3(256), 0, stream, xf, ipw, ipb, q, kbuf);
  hipLaunchKernelGGL(k_attn, dim3(TN_), dim3(256), 0, stream, q, kbuf, attn);
  hipLaunchKernelGGL(k_topk, dim3(TN_), dim3(256), 0, stream, attn, atkb);
  // H1 = tanh(relu(attn_topk @ W1^T + b1))  -- 4096^3 bf16 WMMA GEMM, 32x32 per wave
  hipLaunchKernelGGL(k_wmma_gemm, dim3(128, 16), dim3(256), 0, stream,
                     atkb, w1b, b1, h1b, (float*)nullptr, TN_, TN_);
  // H = tanh(relu(H1 @ W2^T + b2))          -- 4096x128 WMMA GEMM -> d_out (4 waves cover N=128)
  hipLaunchKernelGGL(k_wmma_gemm, dim3(128, 1), dim3(128), 0, stream,
                     h1b, w2b, b2, (__bf16*)nullptr, Hout, E_, TN_);
  hipLaunchKernelGGL(k_lnY, dim3(TN_), dim3(64), 0, stream, xf, lng, lnb, P, Y);
  hipLaunchKernelGGL(k_Z,   dim3(E_),  dim3(64), 0, stream, Hout, Y, Z);
  hipLaunchKernelGGL(k_final, dim3(1024), dim3(256), 0, stream, Hout, Z, xf, xout);
}
